// Method_GCN_58471684768394
// MI455X (gfx1250) — matrix operand order repaired and checked by reading, not compiled, not run on hardware
//
#include <hip/hip_runtime.h>

#define N_NODES 50000
#define N_EDGES 800000
#define IN_DIM  512
#define HID_DIM 128
#define OUT_DIM 64

typedef __attribute__((ext_vector_type(16))) __bf16        v16bf;
typedef __attribute__((ext_vector_type(8)))  float         v8f;
typedef __attribute__((ext_vector_type(4)))  unsigned int  v4u;

union Frag16 { v4u q[2]; v16bf bf; };
union Pack4  { __bf16 b[4]; uint2 u2; };

__device__ __forceinline__ __bf16 f2bf(float f) {
    union { float f; unsigned u; } in; in.f = f;
    unsigned r = in.u + 0x7FFFu + ((in.u >> 16) & 1u);   // round-to-nearest-even
    unsigned short h = (unsigned short)(r >> 16);
    __bf16 o; __builtin_memcpy(&o, &h, 2); return o;
}

// ---- pack W1 [512][128] f32 -> W1^T [128][512] bf16 ------------------------
__global__ __launch_bounds__(256) void gcn_pack_w1(const float* __restrict__ W1,
                                                   __bf16* __restrict__ w1t) {
    int idx = blockIdx.x * 256 + threadIdx.x;      // 65536 total
    int k = idx & (IN_DIM - 1);
    int n = idx >> 9;
    w1t[idx] = f2bf(W1[(size_t)k * HID_DIM + n]);
}

// ---- pack W2 [128][64] f32 -> W2^T [64][128] bf16 --------------------------
__global__ __launch_bounds__(256) void gcn_pack_w2(const float* __restrict__ W2,
                                                   __bf16* __restrict__ w2t) {
    int idx = blockIdx.x * 256 + threadIdx.x;      // 8192 total
    int k = idx & (HID_DIM - 1);
    int n = idx >> 7;
    w2t[idx] = f2bf(W2[(size_t)k * OUT_DIM + n]);
}

// ---- h[i][j] = b1[j]  (seed bias before atomic scatter) --------------------
__global__ __launch_bounds__(256) void gcn_init_h(const float* __restrict__ b1,
                                                  float* __restrict__ h) {
    int idx = blockIdx.x * 256 + threadIdx.x;      // 6.4M total
    h[idx] = __ldg(b1 + (idx & (HID_DIM - 1)));
}

// ---- out[i][j] = b2[j] -----------------------------------------------------
__global__ __launch_bounds__(256) void gcn_init_out(const float* __restrict__ b2,
                                                    float* __restrict__ out) {
    int idx = blockIdx.x * 256 + threadIdx.x;      // 3.2M total
    out[idx] = __ldg(b2 + (idx & (OUT_DIM - 1)));
}

// ---- GEMM1: h0 = x @ W1   (bf16 WMMA, f32 accumulate) ----------------------
// block = 256 (8 waves); one 16-row M-tile per block; wave w -> N-tile w*16.
__global__ __launch_bounds__(256) void gcn_gemm1(const float*  __restrict__ x,
                                                 const __bf16* __restrict__ w1t,
                                                 float*        __restrict__ h0) {
    __shared__ __bf16 sX[16 * IN_DIM];             // 16 KB, bf16 x-strip
    const int m0  = blockIdx.x * 16;
    const int tid = threadIdx.x;

    // cooperative load+convert: 16x512 f32 -> bf16 LDS
    const float4* src = (const float4*)(x + (size_t)m0 * IN_DIM);
    uint2* sXp = (uint2*)sX;
#pragma unroll
    for (int i = 0; i < 8; ++i) {
        int idx = tid + i * 256;                   // 0..2047 float4 chunks
        float4 v = src[idx];
        Pack4 p;
        p.b[0] = f2bf(v.x); p.b[1] = f2bf(v.y);
        p.b[2] = f2bf(v.z); p.b[3] = f2bf(v.w);
        sXp[idx] = p.u2;
    }
    __syncthreads();

    const int wave = tid >> 5, lane = tid & 31;
    const int half = lane >> 4, l16 = lane & 15;
    const int n0   = wave * 16;

    // A frag: row m=l16, two 8-elem K runs at k0+half*8 and k0+16+half*8
    const __bf16* abase = sX + l16 * IN_DIM + half * 8;
    // B frag: col n=l16 of this N-tile, two 8-elem K runs at k0+half*8 and k0+16+half*8
    const __bf16* bbase = w1t + (size_t)(n0 + l16) * IN_DIM + half * 8;

    v8f acc = {};
#pragma unroll 4
    for (int k0 = 0; k0 < IN_DIM; k0 += 32) {
        __builtin_prefetch(bbase + k0 + 128, 0, 0);   // global_prefetch_b8
        Frag16 fa, fb;
        fa.q[0] = *(const v4u*)(abase + k0);
        fa.q[1] = *(const v4u*)(abase + k0 + 16);
        fb.q[0] = *(const v4u*)(bbase + k0);
        fb.q[1] = *(const v4u*)(bbase + k0 + 16);
        acc = __builtin_amdgcn_wmma_f32_16x16x32_bf16(false, fa.bf, false, fb.bf,
                                                      (short)0, acc, false, false);
    }
    // C/D layout: VGPR r -> M = r + half*8, N = l16
    float* dst = h0 + (size_t)(m0 + half * 8) * HID_DIM + n0 + l16;
#pragma unroll
    for (int r = 0; r < 8; ++r) dst[r * HID_DIM] = acc[r];
}

// ---- SPMM1: h[row] += val * h0[col]  (one wave32 per edge, float4/lane) ----
__global__ __launch_bounds__(256) void gcn_spmm1(const int*   __restrict__ erow,
                                                 const int*   __restrict__ ecol,
                                                 const float* __restrict__ eval,
                                                 const float* __restrict__ h0,
                                                 float*       __restrict__ h) {
    const int e    = blockIdx.x * 8 + (threadIdx.x >> 5);  // grid 100000
    const int lane = threadIdx.x & 31;
    const int r = erow[e], c = ecol[e];
    const float v = eval[e];
    float4 m = ((const float4*)(h0 + (size_t)c * HID_DIM))[lane];
    float* dst = h + (size_t)r * HID_DIM + lane * 4;
    atomicAdd(dst + 0, v * m.x);
    atomicAdd(dst + 1, v * m.y);
    atomicAdd(dst + 2, v * m.z);
    atomicAdd(dst + 3, v * m.w);
}

// ---- relu + dropout(p=0.5, precomputed noise) -> bf16 ----------------------
__global__ __launch_bounds__(256) void gcn_act(const float* __restrict__ h,
                                               const float* __restrict__ drop_u,
                                               __bf16*      __restrict__ hb) {
    int idx = blockIdx.x * 256 + threadIdx.x;      // 6.4M total
    float v = h[idx];
    v = v > 0.0f ? v : 0.0f;
    v = (drop_u[idx] >= 0.5f) ? v * 2.0f : 0.0f;
    hb[idx] = f2bf(v);
}

// ---- GEMM2: h2 = h_drop @ W2  (bf16 WMMA, K=128, N=64) ---------------------
// block = 128 (4 waves); one 16-row M-tile; wave w -> N-tile w*16.
__global__ __launch_bounds__(128) void gcn_gemm2(const __bf16* __restrict__ hb,
                                                 const __bf16* __restrict__ w2t,
                                                 float*        __restrict__ h2) {
    __shared__ __bf16 sH[16 * HID_DIM];            // 4 KB
    const int m0  = blockIdx.x * 16;
    const int tid = threadIdx.x;

    const v4u* src = (const v4u*)(hb + (size_t)m0 * HID_DIM);  // 256 x 16B
    v4u* dstS = (v4u*)sH;
    dstS[tid]       = src[tid];
    dstS[tid + 128] = src[tid + 128];
    __syncthreads();

    const int wave = tid >> 5, lane = tid & 31;
    const int half = lane >> 4, l16 = lane & 15;
    const int n0   = wave * 16;

    const __bf16* abase = sH + l16 * HID_DIM + half * 8;
    const __bf16* bbase = w2t + (size_t)(n0 + l16) * HID_DIM + half * 8;

    v8f acc = {};
#pragma unroll
    for (int k0 = 0; k0 < HID_DIM; k0 += 32) {
        Frag16 fa, fb;
        fa.q[0] = *(const v4u*)(abase + k0);
        fa.q[1] = *(const v4u*)(abase + k0 + 16);
        fb.q[0] = *(const v4u*)(bbase + k0);
        fb.q[1] = *(const v4u*)(bbase + k0 + 16);
        acc = __builtin_amdgcn_wmma_f32_16x16x32_bf16(false, fa.bf, false, fb.bf,
                                                      (short)0, acc, false, false);
    }
    float* dst = h2 + (size_t)(m0 + half * 8) * OUT_DIM + n0 + l16;
#pragma unroll
    for (int r = 0; r < 8; ++r) dst[r * OUT_DIM] = acc[r];
}

// ---- SPMM2: out[row] += val * h2[col]  (wave per edge, float2/lane) --------
__global__ __launch_bounds__(256) void gcn_spmm2(const int*   __restrict__ erow,
                                                 const int*   __restrict__ ecol,
                                                 const float* __restrict__ eval,
                                                 const float* __restrict__ h2,
                                                 float*       __restrict__ out) {
    const int e    = blockIdx.x * 8 + (threadIdx.x >> 5);  // grid 100000
    const int lane = threadIdx.x & 31;
    const int r = erow[e], c = ecol[e];
    const float v = eval[e];
    float2 m = ((const float2*)(h2 + (size_t)c * OUT_DIM))[lane];
    float* dst = out + (size_t)r * OUT_DIM + lane * 2;
    atomicAdd(dst + 0, v * m.x);
    atomicAdd(dst + 1, v * m.y);
}

extern "C" void kernel_launch(void* const* d_in, const int* in_sizes, int n_in,
                              void* d_out, int out_size, void* d_ws, size_t ws_size,
                              hipStream_t stream) {
    (void)in_sizes; (void)n_in; (void)out_size; (void)ws_size;
    const float* x      = (const float*)d_in[0];
    const int*   erow   = (const int*)  d_in[1];
    const int*   ecol   = (const int*)  d_in[2];
    const float* eval   = (const float*)d_in[3];
    const float* drop_u = (const float*)d_in[4];
    const float* W1     = (const float*)d_in[5];
    const float* b1     = (const float*)d_in[6];
    const float* W2     = (const float*)d_in[7];
    const float* b2     = (const float*)d_in[8];
    float* out = (float*)d_out;

    char* ws = (char*)d_ws;
    float*  h0  = (float*) (ws);                    // 25,600,000 B
    float*  h   = (float*) (ws + 25600000);         // 25,600,000 B
    __bf16* hb  = (__bf16*)(ws + 51200000);         // 12,800,000 B
    __bf16* w1t = (__bf16*)(ws + 64000000);         //    131,072 B
    __bf16* w2t = (__bf16*)(ws + 64131072);         //     16,384 B
    float*  h2  = (float*) (ws);                    // alias h0 (dead after spmm1)

    gcn_pack_w1 <<<256,   256, 0, stream>>>(W1, w1t);
    gcn_pack_w2 <<<32,    256, 0, stream>>>(W2, w2t);
    gcn_init_h  <<<25000, 256, 0, stream>>>(b1, h);
    gcn_gemm1   <<<3125,  256, 0, stream>>>(x, w1t, h0);
    gcn_spmm1   <<<100000,256, 0, stream>>>(erow, ecol, eval, h0, h);
    gcn_act     <<<25000, 256, 0, stream>>>(h, drop_u, hb);
    gcn_gemm2   <<<3125,  128, 0, stream>>>(hb, w2t, h2);
    gcn_init_out<<<12500, 256, 0, stream>>>(b2, out);
    gcn_spmm2   <<<100000,256, 0, stream>>>(erow, ecol, eval, h2, out);
}
